// SO3InteractionBlock_28432683499857
// MI455X (gfx1250) — matrix operand order repaired and checked by reading, not compiled, not run on hardware
//
#include <hip/hip_runtime.h>

// SO3 interaction block for MI455X (gfx1250, wave32, WMMA + TDM).
// Edge path restructured as static-weight GEMMs over outer-product features:
//   P[e, kb*16+u] = radial[e,kb] * hs_src[e,u]   (K=128)  @ B_P (128x32)
//   Q_i[e, kb*8+u] = radial[e,kb] * hv_src[e,u,i] (K=64)  @ B_Q ( 64x32)
// All scale constants folded into the prepacked f16 B panels, which are
// staged into LDS once per block with tensor_load_to_lds (TDM).
// A-fragment packing is fully select-free: laneHi-dependence is resolved once
// into tiny register permutations (hsel / rq), so each fragment is exactly
// 8x v_pk_mul_f16 feeding v_wmma_f32_16x16x32_f16.

typedef __attribute__((ext_vector_type(16))) _Float16 v16h;
typedef __attribute__((ext_vector_type(2)))  _Float16 h2;
typedef __attribute__((ext_vector_type(8)))  float    v8f;
typedef __attribute__((ext_vector_type(4)))  unsigned int v4u;
typedef __attribute__((ext_vector_type(8)))  unsigned int v8u;

struct H16 { h2 p[8]; };   // bit_cast carrier for a v16h fragment

constexpr int KM0  = 16;
constexpr int KM1  = 8;
constexpr int KDIM = 40;   // M0 + 3*M1
constexpr int KNB  = 8;    // N_BASIS

// B panel buffer: 12 fragments (8 for P, 4 for Q), each 32 lanes x 16 halves,
// stored fragment-major and lane-major so a lane reads 32 contiguous bytes.
constexpr int NFRAG    = 12;
constexpr int FRAG_ELT = 512;              // 32 lanes * 16 halves
constexpr int BELT     = NFRAG * FRAG_ELT; // 6144 halves = 12 KB

// ---------------------------------------------------------------------------
// Prep kernel: fold Wfc slices + all scale constants into f16 B panels,
// already swizzled into per-fragment WMMA lane layout:
//   fragment f, lane l, element e (j = e>>1, hh = e&1):
//     K = kk(f) + ((l>=16)?8:0) + (e&7) + ((e>=8)?16:0),  N = (l&15) + nb(f)*16
// P fragments f=0..7: kk = (f>>1)*32, nb = f&1,  K -> (kb = K>>4, u = K&15)
// Q fragments f=8..11: kk = ((f-8)>>1)*32, nb = (f-8)&1, K -> (kb=K>>3,u=K&7)
// Scales (fold 1/sqrt(N_BASIS), channel norms and inv2):
//   w00,w01: 1/16      w11: 1/sqrt(384)      w10: 1/(8*sqrt(2))
// ---------------------------------------------------------------------------
__global__ void prep_weights_kernel(const float* __restrict__ Wfc,
                                    _Float16* __restrict__ Bpk) {
  const int tid    = blockIdx.x * blockDim.x + threadIdx.x;
  const int stride = blockDim.x * gridDim.x;
  const float sP  = 1.0f / 16.0f;
  const float sQs = 0.05103103630798288f;   // 1/sqrt(384)
  const float sQv = 0.08838834764831845f;   // 1/(8*sqrt(2))
  for (int idx = tid; idx < BELT; idx += stride) {
    const int f    = idx >> 9;
    const int r    = idx & 511;
    const int lane = r >> 4;
    const int elem = r & 15;
    const int j    = elem >> 1;
    const int hh   = elem & 1;
    const int khi  = (lane >> 4) << 4;
    const int Ncol = lane & 15;
    float v = 0.0f;
    if (f < 8) {
      const int kk = (f >> 1) * 32, nb = f & 1;
      const int K = kk + (khi >> 1) + (elem & 7) + ((elem >> 3) << 4);       // 0..127
      const int N = Ncol + nb * 16;
      const int kb = K >> 4, u = K & 15;
      if (N < 16)      v = Wfc[kb * 576 + u * 16 + N] * sP;             // w00
      else if (N < 24) v = Wfc[kb * 576 + 256 + u * 8 + (N - 16)] * sP;  // w01
    } else {
      const int fq = f - 8;
      const int kk = (fq >> 1) * 32, nb = fq & 1;
      const int K = kk + (khi >> 1) + (elem & 7) + ((elem >> 3) << 4);       // 0..63
      const int N = Ncol + nb * 16;
      const int kb = K >> 3, u = K & 7;
      if (N < 16)      v = Wfc[kb * 576 + 448 + u * 16 + N] * sQs;       // w11
      else if (N < 24) v = Wfc[kb * 576 + 384 + u * 8 + (N - 16)] * sQv;  // w10
    }
    Bpk[idx] = (_Float16)v;
  }
}

// ---------------------------------------------------------------------------
// Node pre kernel: h = lin1(x)
// ---------------------------------------------------------------------------
__global__ __launch_bounds__(256) void node_pre_kernel(
    const float* __restrict__ x, const float* __restrict__ W1s,
    const float* __restrict__ W1v, float* __restrict__ h, int nNodes) {
  __shared__ float sWs[256];
  __shared__ float sWv[64];
  const int tid = threadIdx.x;
  if (tid < 256) sWs[tid] = W1s[tid];
  if (tid < 64)  sWv[tid] = W1v[tid];
  __syncthreads();
  const int node = blockIdx.x * 256 + tid;
  if (node >= nNodes) return;
  const float* xr = x + (size_t)node * KDIM;
  float out[KDIM];
  const float s0 = 0.25f;                   // 1/sqrt(16)
  const float s1 = 0.35355339059327373f;    // 1/sqrt(8)
  for (int w = 0; w < KM0; ++w) {
    float acc = 0.0f;
    for (int u = 0; u < KM0; ++u) acc += xr[u] * sWs[u * 16 + w];
    out[w] = acc * s0;
  }
  for (int w = 0; w < KM1; ++w)
    for (int i = 0; i < 3; ++i) {
      float acc = 0.0f;
      for (int u = 0; u < KM1; ++u) acc += xr[16 + u * 3 + i] * sWv[u * 8 + w];
      out[16 + w * 3 + i] = acc * s1;
    }
  float* hr = h + (size_t)node * KDIM;
  for (int j = 0; j < KDIM; ++j) hr[j] = out[j];
}

// ---------------------------------------------------------------------------
// Edge message kernel: one 16-edge tile per wave, 8 waves/block.
// B panels arrive in LDS via the Tensor Data Mover; A fragments are packed
// with v_pk_mul_f16 from register-cached row data; B fragments are single
// 32-byte LDS reads. 20 v_wmma_f32_16x16x32_f16 per 16 edges.
// ---------------------------------------------------------------------------
__global__ __launch_bounds__(256) void edge_msg_kernel(
    const float* __restrict__ edge_radial, const float* __restrict__ edge_sh,
    const int* __restrict__ edge_src, const int* __restrict__ edge_dst,
    const float* __restrict__ h, const _Float16* __restrict__ Bpk,
    float* __restrict__ accum, int nEdges, int nTiles) {
  __shared__ __align__(32) _Float16 sB[BELT];

  const int tid  = threadIdx.x;
  const int wave = tid >> 5;
  const int lane = tid & 31;

  // ---- TDM: stage the 12 KB B panel block into LDS (wave 0 issues) ----
  if (wave == 0) {
    const unsigned long long ga = (unsigned long long)Bpk;
    const unsigned int ldsOff   = (unsigned int)(unsigned long long)(uintptr_t)&sB[0];
    const unsigned int NE = (unsigned int)BELT;  // elements (data_size = 2B)
    v4u g0;
    g0[0] = 1u;                                           // count=1, load D#
    g0[1] = ldsOff;                                       // lds_addr
    g0[2] = (unsigned int)ga;                             // global_addr[31:0]
    g0[3] = (unsigned int)((ga >> 32) & 0x01FFFFFFu) | (2u << 30); // [56:32]|type=2
    v8u g1;
    g1[0] = (1u << 16);            // workgroup_mask=0, data_size=1 (2 bytes)
    g1[1] = (NE & 0xFFFFu) << 16;  // tensor_dim0[15:0]
    g1[2] = (NE >> 16) | (1u << 16); // tensor_dim0[31:16] | tensor_dim1=1
    g1[3] = (NE << 16);            // tensor_dim1[31:16]=0 | tile_dim0
    g1[4] = 0u;                    // tile_dim1=0 (unused), tile_dim2=0
    g1[5] = NE;                    // tensor_dim0_stride[31:0]
    g1[6] = 0u;
    g1[7] = 0u;
    asm volatile("tensor_load_to_lds %0, %1" : : "s"(g0), "s"(g1) : "memory");
    __builtin_amdgcn_s_wait_tensorcnt(0);
  }
  __syncthreads();

  const int tile = blockIdx.x * 8 + wave;
  if (tile >= nTiles) return;
  const int e0     = tile * 16;
  const int laneHi = lane >> 4;
  const int m      = lane & 15;   // A row handled by this lane
  const int n      = m;           // C/D column held by this lane

  // ---- per-lane A-row data, converted once to packed f16 ----
  const int  eA    = e0 + m;
  const bool okA   = eA < nEdges;
  const int  esA   = okA ? eA : 0;
  float4 ra = *(const float4*)&edge_radial[(size_t)esA * KNB];
  float4 rb = *(const float4*)&edge_radial[(size_t)esA * KNB + 4];
  if (!okA) { ra.x = ra.y = ra.z = ra.w = 0.0f; rb = ra; }
  h2 rp[8];
  rp[0] = h2{(_Float16)ra.x, (_Float16)ra.x};
  rp[1] = h2{(_Float16)ra.y, (_Float16)ra.y};
  rp[2] = h2{(_Float16)ra.z, (_Float16)ra.z};
  rp[3] = h2{(_Float16)ra.w, (_Float16)ra.w};
  rp[4] = h2{(_Float16)rb.x, (_Float16)rb.x};
  rp[5] = h2{(_Float16)rb.y, (_Float16)rb.y};
  rp[6] = h2{(_Float16)rb.z, (_Float16)rb.z};
  rp[7] = h2{(_Float16)rb.w, (_Float16)rb.w};

  const int src = edge_src[esA];
  const float* hrow = h + (size_t)src * KDIM;
  float hf[KDIM];
#pragma unroll
  for (int q = 0; q < KDIM / 4; ++q) {
    const float4 t = *(const float4*)&hrow[q * 4];
    hf[q * 4 + 0] = t.x; hf[q * 4 + 1] = t.y;
    hf[q * 4 + 2] = t.z; hf[q * 4 + 3] = t.w;
  }
  h2 hs_pk[8];
#pragma unroll
  for (int p = 0; p < 8; ++p)
    hs_pk[p] = h2{(_Float16)hf[2 * p], (_Float16)hf[2 * p + 1]};
  h2 hv_pk[3][4];
#pragma unroll
  for (int i = 0; i < 3; ++i)
#pragma unroll
    for (int p = 0; p < 4; ++p)
      hv_pk[i][p] = h2{(_Float16)hf[16 + 6 * p + i], (_Float16)hf[16 + 6 * p + 3 + i]};

  // ---- resolve laneHi-dependence once (select-free fragment packing) ----
  // P-GEMM hs pair index: (vg&3) + 4*laneHi  ->  hsel[vg&3]
  h2 hsel[4];
#pragma unroll
  for (int c = 0; c < 4; ++c) hsel[c] = hs_pk[4 * laneHi + c];
  // Q-GEMM radial index: 2*(kk/16 + (vg<4?0:1)) + laneHi  ->  rq[kk/16 + (vg<4?0:1)]
  h2 rq[4];
#pragma unroll
  for (int c = 0; c < 4; ++c) rq[c] = rp[2 * c + laneHi];

  // ---- combine-phase per-row data (rows r + laneHi*8) ----
  float sh_s[8], sh_v[8][3];
  int dstR[8];
#pragma unroll
  for (int r = 0; r < 8; ++r) {
    const int  er = e0 + r + laneHi * 8;
    const bool ok = er < nEdges;
    const int  ec = ok ? er : 0;
    float4 q = *(const float4*)&edge_sh[(size_t)ec * 4];
    if (!ok) { q.x = q.y = q.z = q.w = 0.0f; }
    sh_s[r] = q.x; sh_v[r][0] = q.y; sh_v[r][1] = q.z; sh_v[r][2] = q.w;
    dstR[r] = ok ? edge_dst[ec] : 0;
  }

  // fragment loader: one 32-byte LDS read per fragment
  const _Float16* sBl = sB;
#define LOAD_FRAG(f) (*(const v16h*)&sBl[((f) * 32 + lane) * 16])

  // ---- GEMM-P: K = 128 ----
  // A element (vg): radial kb = kk/16 + (vg<4?0:1) (compile-time),
  //                 hs pair  = hsel[vg&3]
  v8f OP0 = {};
  v8f OP1 = {};
#pragma unroll
  for (int kk = 0; kk < 128; kk += 32) {
    H16 t;
#pragma unroll
    for (int vg = 0; vg < 8; ++vg) {
      const int kb = (kk >> 4) + (vg < 4 ? 0 : 1);
      t.p[vg] = rp[kb] * hsel[vg & 3];
    }
    const v16h a  = __builtin_bit_cast(v16h, t);
    const v16h b0 = LOAD_FRAG((kk >> 5) * 2 + 0);
    const v16h b1 = LOAD_FRAG((kk >> 5) * 2 + 1);
    OP0 = __builtin_amdgcn_wmma_f32_16x16x32_f16(false, a, false, b0, (short)0, OP0, false, false);
    OP1 = __builtin_amdgcn_wmma_f32_16x16x32_f16(false, a, false, b1, (short)0, OP1, false, false);
  }

  float msgS[8];
#pragma unroll
  for (int r = 0; r < 8; ++r) msgS[r] = sh_s[r] * OP0[r];

  // hoist the 4 Q-panel B fragments (shared across all 3 components)
  const v16h qb00 = LOAD_FRAG(8);
  const v16h qb01 = LOAD_FRAG(9);
  const v16h qb10 = LOAD_FRAG(10);
  const v16h qb11 = LOAD_FRAG(11);

  // ---- GEMM-Q per vector component i: K = 64 ----
  // A element (vg): radial = rq[kk/16 + (vg<4?0:1)], hv pair = hv_pk[i][vg&3]
#pragma unroll
  for (int i = 0; i < 3; ++i) {
    v8f Q0 = {};
    v8f Q1 = {};
#pragma unroll
    for (int kk = 0; kk < 64; kk += 32) {
      H16 t;
#pragma unroll
      for (int vg = 0; vg < 8; ++vg) {
        const int rqi = (kk >> 5) * 2 + (vg < 4 ? 0 : 1);
        t.p[vg] = rq[rqi] * hv_pk[i][vg & 3];
      }
      const v16h a  = __builtin_bit_cast(v16h, t);
      const v16h b0 = (kk == 0) ? qb00 : qb10;
      const v16h b1 = (kk == 0) ? qb01 : qb11;
      Q0 = __builtin_amdgcn_wmma_f32_16x16x32_f16(false, a, false, b0, (short)0, Q0, false, false);
      Q1 = __builtin_amdgcn_wmma_f32_16x16x32_f16(false, a, false, b1, (short)0, Q1, false, false);
    }
#pragma unroll
    for (int r = 0; r < 8; ++r) {
      msgS[r] += sh_v[r][i] * Q0[r];
      if (n < KM1) {
        const float mv = sh_v[r][i] * OP1[r] + sh_s[r] * Q1[r];
        __hip_atomic_fetch_add(&accum[(size_t)dstR[r] * KDIM + KM0 + n * 3 + i], mv,
                               __ATOMIC_RELAXED, __HIP_MEMORY_SCOPE_AGENT);
      }
    }
  }
#pragma unroll
  for (int r = 0; r < 8; ++r) {
    __hip_atomic_fetch_add(&accum[(size_t)dstR[r] * KDIM + n], msgS[r],
                           __ATOMIC_RELAXED, __HIP_MEMORY_SCOPE_AGENT);
  }
#undef LOAD_FRAG
}

// ---------------------------------------------------------------------------
// Node post kernel: deg-normalize, lin2, gated softplus, + self interaction
// ---------------------------------------------------------------------------
__device__ __forceinline__ float ssp(float v) {
  const float sp = (v > 20.0f) ? v : log1pf(expf(v));
  return sp - 0.6931471805599453f;
}

__global__ __launch_bounds__(256) void node_post_kernel(
    const float* __restrict__ x, const float* __restrict__ accum,
    const float* __restrict__ W2s, const float* __restrict__ W2v,
    const float* __restrict__ R00, const float* __restrict__ R01,
    const float* __restrict__ R10, const float* __restrict__ R11,
    float* __restrict__ out, int nNodes) {
  __shared__ float sR00[4096];
  __shared__ float sR01[1024];
  __shared__ float sR10[1024];
  __shared__ float sR11[1024];
  __shared__ float sW2s[256];
  __shared__ float sW2v[64];
  const int tid = threadIdx.x;
  for (int i = tid; i < 4096; i += 256) sR00[i] = R00[i];
  for (int i = tid; i < 1024; i += 256) { sR01[i] = R01[i]; sR10[i] = R10[i]; sR11[i] = R11[i]; }
  if (tid < 256) sW2s[tid] = W2s[tid];
  if (tid < 64)  sW2v[tid] = W2v[tid];
  __syncthreads();
  const int node = blockIdx.x * 256 + tid;
  if (node >= nNodes) return;

  // accumulated messages / deg (deg = sqrt(E/N) = 4)
  const float* ac = accum + (size_t)node * KDIM;
  float as[16], av[8][3];
  for (int u = 0; u < 16; ++u) as[u] = ac[u] * 0.25f;
  for (int u = 0; u < 8; ++u)
    for (int i = 0; i < 3; ++i) av[u][i] = ac[16 + u * 3 + i] * 0.25f;

  // lin2
  float hs[16], hv[8][3];
  for (int w = 0; w < 16; ++w) {
    float a = 0.0f;
    for (int u = 0; u < 16; ++u) a += as[u] * sW2s[u * 16 + w];
    hs[w] = a * 0.25f;
  }
  for (int w = 0; w < 8; ++w)
    for (int i = 0; i < 3; ++i) {
      float a = 0.0f;
      for (int u = 0; u < 8; ++u) a += av[u][i] * sW2v[u * 8 + w];
      hv[w][i] = a * 0.35355339059327373f;
    }

  // gated softplus
  const float eps2 = 1e-16f;
  for (int w = 0; w < 16; ++w) {
    const float ns = sqrtf(hs[w] * hs[w] + eps2);
    hs[w] = hs[w] / ns * ssp(ns);
  }
  for (int w = 0; w < 8; ++w) {
    const float nv = sqrtf(hv[w][0] * hv[w][0] + hv[w][1] * hv[w][1] +
                           hv[w][2] * hv[w][2] + eps2);
    const float g = ssp(nv) / nv;
    hv[w][0] *= g; hv[w][1] *= g; hv[w][2] *= g;
  }

  // self interaction with original x
  const float* xr = x + (size_t)node * KDIM;
  float xs[16], xv[8][3];
  for (int u = 0; u < 16; ++u) xs[u] = xr[u];
  for (int u = 0; u < 8; ++u)
    for (int i = 0; i < 3; ++i) xv[u][i] = xr[16 + u * 3 + i];

  const float inv2 = 0.7071067811865476f;
  const float c00 = inv2 / 16.0f;
  const float c11 = inv2 * 0.07216878364870323f;  // 1/sqrt(192)
  const float c01 = inv2 * 0.08838834764831845f;  // 1/sqrt(128)

  float* o = out + (size_t)node * KDIM;
  for (int w = 0; w < 16; ++w) {
    float a = 0.0f;
    for (int u = 0; u < 16; ++u) {
      const float xu = xs[u];
      for (int v = 0; v < 16; ++v) a += xu * xs[v] * sR00[u * 256 + v * 16 + w];
    }
    float b = 0.0f;
    for (int u = 0; u < 8; ++u)
      for (int v = 0; v < 8; ++v) {
        const float dot = xv[u][0] * xv[v][0] + xv[u][1] * xv[v][1] + xv[u][2] * xv[v][2];
        b += dot * sR11[u * 128 + v * 16 + w];
      }
    o[w] = hs[w] + a * c00 + b * c11;
  }
  for (int w = 0; w < 8; ++w)
    for (int i = 0; i < 3; ++i) {
      float a = 0.0f;
      for (int u = 0; u < 16; ++u)
        for (int v = 0; v < 8; ++v) a += xs[u] * xv[v][i] * sR01[u * 64 + v * 8 + w];
      float b = 0.0f;
      for (int u = 0; u < 8; ++u)
        for (int v = 0; v < 16; ++v) b += xv[u][i] * xs[v] * sR10[u * 128 + v * 8 + w];
      o[16 + w * 3 + i] = hv[w][i] + (a + b) * c01;
    }
}

// ---------------------------------------------------------------------------
// Launcher
// ---------------------------------------------------------------------------
extern "C" void kernel_launch(void* const* d_in, const int* in_sizes, int n_in,
                              void* d_out, int out_size, void* d_ws, size_t ws_size,
                              hipStream_t stream) {
  (void)n_in; (void)out_size; (void)ws_size;
  const float* x           = (const float*)d_in[0];
  const float* edge_radial = (const float*)d_in[1];
  const float* edge_sh     = (const float*)d_in[2];
  const int*   edge_src    = (const int*)d_in[3];
  const int*   edge_dst    = (const int*)d_in[4];
  // d_in[5] = n_nodes (device scalar) -- derived on host from in_sizes instead
  const float* W1s = (const float*)d_in[6];
  const float* W1v = (const float*)d_in[7];
  const float* Wfc = (const float*)d_in[8];
  const float* W2s = (const float*)d_in[9];
  const float* W2v = (const float*)d_in[10];
  const float* R00 = (const float*)d_in[11];
  const float* R01 = (const float*)d_in[12];
  const float* R10 = (const float*)d_in[13];
  const float* R11 = (const float*)d_in[14];
  float* out = (float*)d_out;

  const int nNodes = in_sizes[0] / KDIM;
  const int nEdges = in_sizes[3];
  const int nTiles = (nEdges + 15) / 16;

  // workspace: h | accum | Bpk (f16, fragment-major)
  float* h     = (float*)d_ws;
  float* accum = h + (size_t)nNodes * KDIM;
  _Float16* Bpk = (_Float16*)(accum + (size_t)nNodes * KDIM);

  hipMemsetAsync(accum, 0, (size_t)nNodes * KDIM * sizeof(float), stream);
  prep_weights_kernel<<<8, 256, 0, stream>>>(Wfc, Bpk);
  node_pre_kernel<<<(nNodes + 255) / 256, 256, 0, stream>>>(x, W1s, W1v, h, nNodes);
  edge_msg_kernel<<<(nTiles + 7) / 8, 256, 0, stream>>>(
      edge_radial, edge_sh, edge_src, edge_dst, h, Bpk, accum, nEdges, nTiles);
  node_post_kernel<<<(nNodes + 255) / 256, 256, 0, stream>>>(
      x, accum, W2s, W2v, R00, R01, R10, R11, out, nNodes);
}
